// NodeRegressor_17952963297293
// MI455X (gfx1250) — matrix operand order repaired and checked by reading, not compiled, not run on hardware
//
#include <hip/hip_runtime.h>
#include <hip/hip_bf16.h>

#define N_NODES 100000
#define N_EDGES 2000000
#define HID 128

typedef _Float16 v16h __attribute__((ext_vector_type(16)));
typedef _Float16 v8h  __attribute__((ext_vector_type(8)));
typedef _Float16 v4h  __attribute__((ext_vector_type(4)));
typedef float    v8f  __attribute__((ext_vector_type(8)));

// ---------------- Edge scatter for layer 1 (x is 6-wide) + degree counts ----
__global__ void k_scatter1(const int* __restrict__ ei, const float* __restrict__ x,
                           float* __restrict__ agg1, float* __restrict__ cnt) {
    int e = blockIdx.x * 256 + threadIdx.x;
    if (e >= N_EDGES) return;
    int s = ei[e];
    int d = ei[N_EDGES + e];
    atomicAdd(&cnt[d], 1.0f);
    const float* xs = x + (size_t)s * 6;
    float* ad = agg1 + (size_t)d * 6;
#pragma unroll
    for (int k = 0; k < 6; ++k) atomicAdd(&ad[k], xs[k]);
}

// ---------------- Layer 1: h1 = relu(mean1 @ W1l + b1l + x @ W1r), K=6 ------
__global__ void k_layer1(const float* __restrict__ x, const float* __restrict__ agg1,
                         const float* __restrict__ cnt,
                         const float* __restrict__ W1l, const float* __restrict__ b1l,
                         const float* __restrict__ W1r, _Float16* __restrict__ h1) {
    int node = blockIdx.x;          // N_NODES blocks
    int c = threadIdx.x;            // 128 threads
    float inv = 1.0f / fmaxf(cnt[node], 1.0f);
    const float* xs = x + (size_t)node * 6;
    const float* ag = agg1 + (size_t)node * 6;
    float s = b1l[c];
#pragma unroll
    for (int k = 0; k < 6; ++k)
        s += (ag[k] * inv) * W1l[k * HID + c] + xs[k] * W1r[k * HID + c];
    h1[(size_t)node * HID + c] = (_Float16)fmaxf(s, 0.0f);
}

// ---------------- Edge scatter of h1 (128-wide): 1 wave per edge ------------
__global__ void k_scatter2(const int* __restrict__ ei, const _Float16* __restrict__ h1,
                           float* __restrict__ agg2) {
    int gid = blockIdx.x * 256 + threadIdx.x;
    int e = gid >> 5;               // one wave32 per edge
    if (e >= N_EDGES) return;
    int lane = threadIdx.x & 31;
    int s = ei[e];
    int d = ei[N_EDGES + e];
    v4h hv = ((const v4h*)(h1 + (size_t)s * HID))[lane];   // 8B = 4 halves
    float* ad = agg2 + (size_t)d * HID + lane * 4;
#pragma unroll
    for (int i = 0; i < 4; ++i) atomicAdd(&ad[i], (float)hv[i]);
}

// ---------------- mean2 = agg2 / max(cnt,1), convert to f16 -----------------
__global__ void k_mean2(const float* __restrict__ agg2, const float* __restrict__ cnt,
                        _Float16* __restrict__ mean2) {
    int gid = blockIdx.x * 256 + threadIdx.x;   // exactly N*128 threads
    int node = gid >> 7;
    float inv = 1.0f / fmaxf(cnt[node], 1.0f);
    mean2[gid] = (_Float16)(agg2[gid] * inv);
}

// ---------------- Transpose + convert weights: Wt[n][k] = (f16)W[k][n] ------
__global__ void k_transW(const float* __restrict__ W, _Float16* __restrict__ Wt) {
    int gid = blockIdx.x * 256 + threadIdx.x;   // 16384 threads
    int n = gid >> 7, k = gid & 127;
    Wt[n * HID + k] = (_Float16)W[k * HID + n];
}

// ---------------- Layer 2 (WMMA) fused with final linear --------------------
// block = 256 (8 waves). Each wave: one 16x16 (MxN) tile. Block covers all 128
// output channels of 16 nodes; fused h2 @ Wlin + blin via shuffle+LDS reduce.
__global__ __launch_bounds__(256) void k_layer2_fused(
    const _Float16* __restrict__ mean2, const _Float16* __restrict__ h1,
    const _Float16* __restrict__ w2lt,  const _Float16* __restrict__ w2rt,
    const float* __restrict__ b2l, const float* __restrict__ wlin,
    const float* __restrict__ blin, float* __restrict__ out) {
    __shared__ float red[8][16];

    int tid  = threadIdx.x;
    int wid  = tid >> 5;
    int lane = tid & 31;
    int r    = lane & 15;          // A row / B column within tile
    int high = lane >> 4;          // half-wave selector
    int m0   = blockIdx.x * 16;    // node tile base (6250 blocks * 16 = 100000)
    int ncol = wid * 16 + r;       // output channel

    // A layout: 16 contiguous K-halves at high*16 (matches B)
    const _Float16* aM = mean2 + (size_t)(m0 + r) * HID;
    const _Float16* aH = h1    + (size_t)(m0 + r) * HID;
    int aoff = high * 16;
    // B layout: column = lane&15; 16 contiguous K-halves at kb2 = high*16
    const _Float16* bLp = w2lt + (size_t)ncol * HID + high * 16;
    const _Float16* bRp = w2rt + (size_t)ncol * HID + high * 16;

    float bias = b2l[ncol];
    v8f c;
#pragma unroll
    for (int j = 0; j < 8; ++j) c[j] = bias;

#pragma unroll
    for (int kt = 0; kt < 4; ++kt) {
        int kb = kt * 32;
        v16h a, b;
        v8h lo = *(const v8h*)(aM + kb + aoff);
        v8h hi = *(const v8h*)(aM + kb + aoff + 8);
#pragma unroll
        for (int i = 0; i < 8; ++i) { a[i] = lo[i]; a[8 + i] = hi[i]; }
        b = *(const v16h*)(bLp + kb);
        c = __builtin_amdgcn_wmma_f32_16x16x32_f16(false, a, false, b,
                                                   (short)0, c, false, false);
        lo = *(const v8h*)(aH + kb + aoff);
        hi = *(const v8h*)(aH + kb + aoff + 8);
#pragma unroll
        for (int i = 0; i < 8; ++i) { a[i] = lo[i]; a[8 + i] = hi[i]; }
        b = *(const v16h*)(bRp + kb);
        c = __builtin_amdgcn_wmma_f32_16x16x32_f16(false, a, false, b,
                                                   (short)0, c, false, false);
    }

    // ReLU + fused (h2 @ Wlin): lane holds rows high*8+j, column ncol.
    float wl = wlin[ncol];
#pragma unroll
    for (int j = 0; j < 8; ++j) {
        float p = fmaxf(c[j], 0.0f) * wl;
        p += __shfl_xor(p, 1);
        p += __shfl_xor(p, 2);
        p += __shfl_xor(p, 4);
        p += __shfl_xor(p, 8);          // sum over 16 columns in half-wave
        if (r == 0) red[wid][high * 8 + j] = p;
    }
    __syncthreads();
    if (tid < 16) {                     // deterministic cross-wave combine
        float s = blin[0];
#pragma unroll
        for (int w = 0; w < 8; ++w) s += red[w][tid];
        out[m0 + tid] = s;
    }
}

extern "C" void kernel_launch(void* const* d_in, const int* in_sizes, int n_in,
                              void* d_out, int out_size, void* d_ws, size_t ws_size,
                              hipStream_t stream) {
    const float* x    = (const float*)d_in[0];
    const int*   ei   = (const int*)d_in[1];      // [2, E]
    const float* W1l  = (const float*)d_in[2];
    const float* b1l  = (const float*)d_in[3];
    const float* W1r  = (const float*)d_in[4];
    const float* W2l  = (const float*)d_in[5];
    const float* b2l  = (const float*)d_in[6];
    const float* W2r  = (const float*)d_in[7];
    const float* Wlin = (const float*)d_in[8];
    const float* blin = (const float*)d_in[9];
    float* out = (float*)d_out;

    // ---- workspace layout (f32 accumulators first so one memset zeros them)
    float* cnt  = (float*)d_ws;                       // N
    float* agg1 = cnt + N_NODES;                      // N*6
    float* agg2 = agg1 + (size_t)N_NODES * 6;         // N*128
    _Float16* h1    = (_Float16*)(agg2 + (size_t)N_NODES * HID);  // N*128 f16
    _Float16* mean2 = h1 + (size_t)N_NODES * HID;                 // N*128 f16
    _Float16* w2lt  = mean2 + (size_t)N_NODES * HID;              // 128*128 f16
    _Float16* w2rt  = w2lt + HID * HID;                           // 128*128 f16

    size_t zero_bytes = (size_t)N_NODES * (1 + 6 + HID) * sizeof(float);
    hipMemsetAsync(d_ws, 0, zero_bytes, stream);

    // weight transpose/convert (independent of graph data)
    k_transW<<<(HID * HID) / 256, 256, 0, stream>>>(W2l, w2lt);
    k_transW<<<(HID * HID) / 256, 256, 0, stream>>>(W2r, w2rt);

    // layer-1 aggregation + transform
    k_scatter1<<<(N_EDGES + 255) / 256, 256, 0, stream>>>(ei, x, agg1, cnt);
    k_layer1<<<N_NODES, HID, 0, stream>>>(x, agg1, cnt, W1l, b1l, W1r, h1);

    // layer-2 aggregation (the bandwidth/atomic bottleneck)
    long long t2 = (long long)N_EDGES * 32;
    k_scatter2<<<(unsigned)((t2 + 255) / 256), 256, 0, stream>>>(ei, h1, agg2);
    k_mean2<<<((size_t)N_NODES * HID) / 256, 256, 0, stream>>>(agg2, cnt, mean2);

    // layer-2 WMMA GEMMs fused with final linear head
    k_layer2_fused<<<N_NODES / 16, 256, 0, stream>>>(mean2, h1, w2lt, w2rt,
                                                     b2l, Wlin, blin, out);
}
